// UnetAttentionBlock_42193758716345
// MI455X (gfx1250) — matrix operand order repaired and checked by reading, not compiled, not run on hardware
//
#include <hip/hip_runtime.h>
#include <hip/hip_bf16.h>
#include <math.h>

typedef __attribute__((ext_vector_type(16))) __bf16 bf16x16;
typedef __attribute__((ext_vector_type(8)))  __bf16 bf16x8;
typedef __attribute__((ext_vector_type(8)))  float  f32x8;

#define CDIM   640
#define GDIM   32
#define NHEAD  8
#define HDDIM  80
#define BDIM   8
#define HWPIX  1024
#define STOK   8192   /* B*HW */
#define PTOK   77
#define LNEPS  1e-5f

// ---------------------------------------------------------------- WMMA helpers

static __device__ inline f32x8 wmma_bf16(bf16x16 a, bf16x16 b, f32x8 c) {
  return __builtin_amdgcn_wmma_f32_16x16x32_bf16(false, a, false, b, (short)0, c,
                                                 false, false);
}

// A-frag (16x32, MxK) from a row-major bf16 tile, row stride ldk halves.
// lane&15 = M row; lanes 0-15 take K {kBase+0..7, kBase+16..23},
// lanes 16-31 take K {kBase+8..15, kBase+24..31}.  Two b128 loads.
static __device__ inline bf16x16 ldsA_frag(const __bf16* base, int ldk,
                                           int mBase, int kBase) {
  int lane = threadIdx.x & 31;
  int m = mBase + (lane & 15);
  int khi = (lane >> 4) << 3;
  const __bf16* p = base + (size_t)m * ldk + kBase + khi;
  bf16x8 lo = *(const bf16x8*)(p);
  bf16x8 hi = *(const bf16x8*)(p + 16);
  return __builtin_shufflevector(lo, hi, 0, 1, 2, 3, 4, 5, 6, 7,
                                 8, 9, 10, 11, 12, 13, 14, 15);
}

// B-frag (32x16, KxN) where the matrix is stored TRANSPOSED (row = N, col = K):
// B[k][n] = base[n*ldk + k].  Two b128 loads per lane: lanes 0-15 take K
// {kBase+0..7, kBase+16..23}, lanes 16-31 take K {kBase+8..15, kBase+24..31}.
// Works for both LDS tiles and global (transposed-weight) pointers.
static __device__ inline bf16x16 bt_frag(const __bf16* base, int ldk,
                                         int kBase, int nBase) {
  int lane = threadIdx.x & 31;
  int n = nBase + (lane & 15);
  int k0 = kBase + ((lane >> 4) << 3);
  const __bf16* p = base + (size_t)n * ldk + k0;
  bf16x8 lo = *(const bf16x8*)(p);
  bf16x8 hi = *(const bf16x8*)(p + 16);
  return __builtin_shufflevector(lo, hi, 0, 1, 2, 3, 4, 5, 6, 7,
                                 8, 9, 10, 11, 12, 13, 14, 15);
}

// ------------------------------------------------------------- small kernels

// Tiled transpose + fp32->bf16: d[n*K + k] = (bf16)s[k*N + n].
__global__ __launch_bounds__(256) void k_cvt_t(const float* __restrict__ s,
                                               __bf16* __restrict__ d,
                                               int K, int N) {
  __shared__ float tile[32][33];
  int kb = blockIdx.x * 32, nb = blockIdx.y * 32;
  int tx = threadIdx.x, ty = threadIdx.y;
#pragma unroll
  for (int r = ty; r < 32; r += 8)
    tile[r][tx] = s[(size_t)(kb + r) * N + nb + tx];
  __syncthreads();
#pragma unroll
  for (int r = ty; r < 32; r += 8)
    d[(size_t)(nb + r) * K + kb + tx] = (__bf16)tile[tx][r];
}

static __device__ inline void block_reduce2(float& s, float& s2) {
  __shared__ float sm1[8], sm2[8];
#pragma unroll
  for (int off = 16; off; off >>= 1) {
    s  += __shfl_xor(s, off, 32);
    s2 += __shfl_xor(s2, off, 32);
  }
  int wid = threadIdx.x >> 5, lane = threadIdx.x & 31;
  if (lane == 0) { sm1[wid] = s; sm2[wid] = s2; }
  __syncthreads();
  int nw = blockDim.x >> 5;
  float a = 0.f, b = 0.f;
  for (int i = 0; i < nw; ++i) { a += sm1[i]; b += sm2[i]; }
  __syncthreads();
  s = a; s2 = b;
}

// GroupNorm over (C/G, H, W); writes token-major (STOK x C) fp32.
__global__ __launch_bounds__(256) void k_groupnorm(
    const float* __restrict__ x, const float* __restrict__ g,
    const float* __restrict__ b, float* __restrict__ outA) {
  const int CPG = CDIM / GDIM;                 // 20
  int bb = blockIdx.x / GDIM, gg = blockIdx.x % GDIM;
  const float* xp = x + ((size_t)bb * CDIM + gg * CPG) * HWPIX;
  float s = 0.f, s2 = 0.f;
  for (int i = threadIdx.x; i < CPG * HWPIX; i += 256) {
    float v = xp[i]; s += v; s2 += v * v;
  }
  block_reduce2(s, s2);
  float mu  = s  * (1.f / (CPG * HWPIX));
  float var = s2 * (1.f / (CPG * HWPIX)) - mu * mu;
  float inv = rsqrtf(var + LNEPS);
  for (int i = threadIdx.x; i < CPG * HWPIX; i += 256) {
    int c = gg * CPG + i / HWPIX;
    int hw = i % HWPIX;
    float v = (xp[i] - mu) * inv * g[c] + b[c];
    outA[((size_t)bb * HWPIX + hw) * CDIM + c] = v;
  }
}

__global__ __launch_bounds__(256) void k_layernorm(
    const float* __restrict__ in, const float* __restrict__ g,
    const float* __restrict__ b, float* __restrict__ out) {
  int row = blockIdx.x;
  const float* p = in + (size_t)row * CDIM;
  float s = 0.f, s2 = 0.f;
  for (int i = threadIdx.x; i < CDIM; i += 256) {
    float v = p[i]; s += v; s2 += v * v;
  }
  block_reduce2(s, s2);
  float mu  = s  * (1.f / CDIM);
  float var = s2 * (1.f / CDIM) - mu * mu;
  float inv = rsqrtf(var + LNEPS);
  for (int i = threadIdx.x; i < CDIM; i += 256)
    out[(size_t)row * CDIM + i] = (p[i] - mu) * inv * g[i] + b[i];
}

// ------------------------------------------------------------- GEMM (WMMA)

// out(MxN) = epilogue( A(fp32 MxK, lda) @ W^T(bf16 NxK, row stride K) + bias )
// B-fragments are read straight from global (weights are L2-resident on the
// 192MB-L2 MI455X); only the A tile (with fp32->bf16 conversion) uses LDS.
// mode 0: +bias       mode 1: +bias +aux (residual, same layout)
// mode 2: geglu: v = acc+bias(gate); out = aux * gelu(v)  (aux may alias out)
// mode 3: final conv: out is NCHW (B,N,1024) += aux (x, same layout)
// Requires: K % 64 == 0, N % 128 == 0.
__global__ __launch_bounds__(256) void k_gemm(
    int mode, const float* __restrict__ A, const __bf16* __restrict__ Wt,
    const float* __restrict__ bias, const float* __restrict__ aux,
    float* __restrict__ out, int M, int N, int K, int lda, int ldo) {
  constexpr int GM = 64, GN = 128, GK = 64;
  __shared__ __bf16 sA[GM * GK];

  const int m0 = blockIdx.x * GM, n0 = blockIdx.y * GN;
  const int tid = threadIdx.x;
  const int wid = tid >> 5, lane = tid & 31;
  const int wm = (wid & 3) * 16;      // wave row offset in tile
  const int wn = (wid >> 2) * 64;     // wave col offset in tile

  f32x8 acc[4];
#pragma unroll
  for (int nb = 0; nb < 4; ++nb)
#pragma unroll
    for (int j = 0; j < 8; ++j) acc[nb][j] = 0.f;

  const int ar = tid >> 2;            // 0..63  (tile row)
  const int ac = (tid & 3) * 16;      // 0,16,32,48

  for (int kk = 0; kk < K; kk += GK) {
    // stage A: 64x64 fp32 -> bf16 (each thread: 16 floats -> two b128 stores)
    {
      int gm = m0 + ar;
      const float* ap = A + (size_t)gm * lda + kk + ac;
      float4 v0, v1, v2, v3;
      if (gm < M) {
        v0 = *(const float4*)(ap);
        v1 = *(const float4*)(ap + 4);
        v2 = *(const float4*)(ap + 8);
        v3 = *(const float4*)(ap + 12);
      } else {
        v0.x=v0.y=v0.z=v0.w=0.f; v1=v0; v2=v0; v3=v0;
      }
      bf16x8 h0, h1;
      h0[0]=(__bf16)v0.x; h0[1]=(__bf16)v0.y; h0[2]=(__bf16)v0.z; h0[3]=(__bf16)v0.w;
      h0[4]=(__bf16)v1.x; h0[5]=(__bf16)v1.y; h0[6]=(__bf16)v1.z; h0[7]=(__bf16)v1.w;
      h1[0]=(__bf16)v2.x; h1[1]=(__bf16)v2.y; h1[2]=(__bf16)v2.z; h1[3]=(__bf16)v2.w;
      h1[4]=(__bf16)v3.x; h1[5]=(__bf16)v3.y; h1[6]=(__bf16)v3.z; h1[7]=(__bf16)v3.w;
      *(bf16x8*)(&sA[ar * GK + ac])     = h0;
      *(bf16x8*)(&sA[ar * GK + ac + 8]) = h1;
      if (kk + GK < K) __builtin_prefetch(ap + GK, 0, 1);
    }
    __syncthreads();
#pragma unroll
    for (int kf = 0; kf < GK; kf += 32) {
      bf16x16 af = ldsA_frag(sA, GK, wm, kf);
#pragma unroll
      for (int nb = 0; nb < 4; ++nb) {
        bf16x16 bfr = bt_frag(Wt, K, kk + kf, n0 + wn + nb * 16);
        acc[nb] = wmma_bf16(af, bfr, acc[nb]);
      }
    }
    __syncthreads();
  }

  // epilogue
#pragma unroll
  for (int nb = 0; nb < 4; ++nb) {
    int gn = n0 + wn + nb * 16 + (lane & 15);
    float bv = bias[gn];
#pragma unroll
    for (int j = 0; j < 8; ++j) {
      int gm = m0 + wm + ((lane >> 4) << 3) + j;
      if (gm >= M) continue;
      float v = acc[nb][j] + bv;
      size_t oi = (size_t)gm * ldo + gn;
      if (mode == 0) {
        out[oi] = v;
      } else if (mode == 1) {
        out[oi] = v + aux[oi];
      } else if (mode == 2) {
        float a = aux[oi];
        float ge = 0.5f * v * (1.f + erff(v * 0.70710678118654752f));
        out[oi] = a * ge;
      } else {
        int bb = gm >> 10, hw = gm & 1023;
        size_t xi = ((size_t)bb * N + gn) * HWPIX + hw;
        out[xi] = v + aux[xi];
      }
    }
  }
}

// --------------------------------------------------------- Flash attention

#define FA_BM 64
#define FA_BN 64
#define FA_D  96   /* HD=80 padded to 3*32 */

__global__ __launch_bounds__(128) void k_flash(
    const float* __restrict__ qbase, const float* __restrict__ kbase,
    const float* __restrict__ vbase, float* __restrict__ obase,
    int Skv, int ldq, int ldk, int ldo, int qTokB, int kTokB, float scale) {
  int bh = blockIdx.y;
  int b = bh >> 3, h = bh & 7;
  int q0 = blockIdx.x * FA_BM;
  int wid = threadIdx.x >> 5, lane = threadIdx.x & 31;

  const float* qp = qbase + (size_t)b * qTokB * ldq + h * HDDIM;
  const float* kp = kbase + (size_t)b * kTokB * ldk + h * HDDIM;
  const float* vp = vbase + (size_t)b * kTokB * ldk + h * HDDIM;
  float* op = obase + (size_t)b * qTokB * ldo + h * HDDIM;

  __shared__ __bf16 smK[FA_BN * FA_D];        // [token][k]   (B^T for Q@K^T)
  __shared__ __bf16 smVT[FA_D * FA_BN];       // [col][token] (B^T for P@V)
  __shared__ __bf16 smP[4][16 * FA_BN];

  // Q fragments (rows q0 + wid*16 .. +15), scale folded in, pad K>=80 with 0.
  int qrow = q0 + wid * 16 + (lane & 15);
  bf16x16 qf[3];
#pragma unroll
  for (int f = 0; f < 3; ++f) {
    bf16x16 r;
#pragma unroll
    for (int i = 0; i < 16; ++i) {
      int k = f * 32 + ((lane >> 4) << 3) + ((i >= 8) ? 16 : 0) + (i & 7);
      float v = (k < HDDIM) ? qp[(size_t)qrow * ldq + k] * scale : 0.f;
      r[i] = (__bf16)v;
    }
    qf[f] = r;
  }

  f32x8 ob[6];
#pragma unroll
  for (int c = 0; c < 6; ++c)
#pragma unroll
    for (int j = 0; j < 8; ++j) ob[c][j] = 0.f;
  float mrow[8], lrow[8];
#pragma unroll
  for (int j = 0; j < 8; ++j) { mrow[j] = -1e30f; lrow[j] = 0.f; }

  int nkb = (Skv + FA_BN - 1) / FA_BN;
  for (int kb = 0; kb < nkb; ++kb) {
    // stage K (row-major, b128 stores) and V (transposed) tiles, zero padded.
    // 64 tokens x 12 chunks of 8 cols; cols 80..95 are pad.
    for (int idx = threadIdx.x; idx < FA_BN * 12; idx += 128) {
      int t = idx / 12, c8 = idx % 12;
      int c0 = c8 * 8;
      int tok = kb * FA_BN + t;
      bf16x8 kv, vv;
      if (c0 < HDDIM && tok < Skv) {
        const float* kr = kp + (size_t)tok * ldk + c0;
        const float* vr = vp + (size_t)tok * ldk + c0;
        float4 a0 = *(const float4*)(kr);
        float4 a1 = *(const float4*)(kr + 4);
        float4 b0 = *(const float4*)(vr);
        float4 b1 = *(const float4*)(vr + 4);
        kv[0]=(__bf16)a0.x; kv[1]=(__bf16)a0.y; kv[2]=(__bf16)a0.z; kv[3]=(__bf16)a0.w;
        kv[4]=(__bf16)a1.x; kv[5]=(__bf16)a1.y; kv[6]=(__bf16)a1.z; kv[7]=(__bf16)a1.w;
        vv[0]=(__bf16)b0.x; vv[1]=(__bf16)b0.y; vv[2]=(__bf16)b0.z; vv[3]=(__bf16)b0.w;
        vv[4]=(__bf16)b1.x; vv[5]=(__bf16)b1.y; vv[6]=(__bf16)b1.z; vv[7]=(__bf16)b1.w;
      } else {
#pragma unroll
        for (int j = 0; j < 8; ++j) { kv[j] = (__bf16)0.f; vv[j] = (__bf16)0.f; }
      }
      *(bf16x8*)(&smK[t * FA_D + c0]) = kv;
#pragma unroll
      for (int j = 0; j < 8; ++j) smVT[(c0 + j) * FA_BN + t] = vv[j];
    }
    __syncthreads();

    // scores: 16x64 per wave = 4 accum frags, K-dim 96 = 3 wmma steps
    f32x8 sc[4];
#pragma unroll
    for (int nb = 0; nb < 4; ++nb)
#pragma unroll
      for (int j = 0; j < 8; ++j) sc[nb][j] = 0.f;
#pragma unroll
    for (int f = 0; f < 3; ++f)
#pragma unroll
      for (int nb = 0; nb < 4; ++nb) {
        bf16x16 bfr = bt_frag(smK, FA_D, f * 32, nb * 16);
        sc[nb] = wmma_bf16(qf[f], bfr, sc[nb]);
      }

    // mask tail + online softmax (rows distributed: j plus lane-half)
    float corr[8];
#pragma unroll
    for (int j = 0; j < 8; ++j) {
      float mx = -1e30f;
#pragma unroll
      for (int nb = 0; nb < 4; ++nb) {
        int col = kb * FA_BN + nb * 16 + (lane & 15);
        float s = (col < Skv) ? sc[nb][j] : -1e30f;
        sc[nb][j] = s;
        mx = fmaxf(mx, s);
      }
#pragma unroll
      for (int off = 8; off; off >>= 1) mx = fmaxf(mx, __shfl_xor(mx, off, 32));
      float mnew = fmaxf(mrow[j], mx);
      corr[j] = __expf(mrow[j] - mnew);
      mrow[j] = mnew;
      float rs = 0.f;
#pragma unroll
      for (int nb = 0; nb < 4; ++nb) {
        float pe = __expf(sc[nb][j] - mnew);
        sc[nb][j] = pe;
        rs += pe;
      }
#pragma unroll
      for (int off = 8; off; off >>= 1) rs += __shfl_xor(rs, off, 32);
      lrow[j] = lrow[j] * corr[j] + rs;
    }
#pragma unroll
    for (int c = 0; c < 6; ++c)
#pragma unroll
      for (int j = 0; j < 8; ++j) ob[c][j] *= corr[j];

    // P -> wave-private LDS (bf16), then P@V
    __bf16* pw = smP[wid];
#pragma unroll
    for (int nb = 0; nb < 4; ++nb)
#pragma unroll
      for (int j = 0; j < 8; ++j) {
        int r = ((lane >> 4) << 3) + j;
        int cc = nb * 16 + (lane & 15);
        pw[r * FA_BN + cc] = (__bf16)sc[nb][j];
      }
#pragma unroll
    for (int f2 = 0; f2 < 2; ++f2) {
      bf16x16 a = ldsA_frag(pw, FA_BN, 0, f2 * 32);
#pragma unroll
      for (int cb = 0; cb < 6; ++cb) {
        bf16x16 bfr = bt_frag(smVT, FA_BN, f2 * 32, cb * 16);
        ob[cb] = wmma_bf16(a, bfr, ob[cb]);
      }
    }
    __syncthreads();
  }

  // write O / l (cols 80..95 are pad: cb==5 skipped)
#pragma unroll
  for (int j = 0; j < 8; ++j) {
    int r = q0 + wid * 16 + ((lane >> 4) << 3) + j;
    float invl = 1.f / lrow[j];
#pragma unroll
    for (int cb = 0; cb < 5; ++cb) {
      int cc = cb * 16 + (lane & 15);
      op[(size_t)r * ldo + cc] = ob[cb][j] * invl;
    }
  }
}

// ---------------------------------------------------------------- launcher

extern "C" void kernel_launch(void* const* d_in, const int* in_sizes, int n_in,
                              void* d_out, int out_size, void* d_ws, size_t ws_size,
                              hipStream_t stream) {
  (void)in_sizes; (void)n_in; (void)out_size; (void)ws_size;
  const float* x    = (const float*)d_in[0];
  const float* p    = (const float*)d_in[1];
  const float* gn_g = (const float*)d_in[2];
  const float* gn_b = (const float*)d_in[3];
  const float* ci_w = (const float*)d_in[4];
  const float* ci_b = (const float*)d_in[5];
  const float* ln1_g= (const float*)d_in[6];
  const float* ln1_b= (const float*)d_in[7];
  const float* qkv_w= (const float*)d_in[8];
  const float* qkv_b= (const float*)d_in[9];
  const float* saw_w= (const float*)d_in[10];
  const float* saw_b= (const float*)d_in[11];
  const float* ln2_g= (const float*)d_in[12];
  const float* ln2_b= (const float*)d_in[13];
  const float* caq_w= (const float*)d_in[14];
  const float* caq_b= (const float*)d_in[15];
  const float* cak_w= (const float*)d_in[16];
  const float* cak_b= (const float*)d_in[17];
  const float* cav_w= (const float*)d_in[18];
  const float* cav_b= (const float*)d_in[19];
  const float* caw_w= (const float*)d_in[20];
  const float* caw_b= (const float*)d_in[21];
  const float* ln3_g= (const float*)d_in[22];
  const float* ln3_b= (const float*)d_in[23];
  const float* g1_w = (const float*)d_in[24];
  const float* g1_b = (const float*)d_in[25];
  const float* g2_w = (const float*)d_in[26];
  const float* g2_b = (const float*)d_in[27];
  const float* co_w = (const float*)d_in[28];
  const float* co_b = (const float*)d_in[29];
  float* out = (float*)d_out;

  char* ws = (char*)d_ws;
  size_t off = 0;
  auto alloc = [&](size_t bytes) -> void* {
    off = (off + 255) & ~(size_t)255;
    void* r = ws + off;
    off += bytes;
    return r;
  };

  // transposed bf16 weights: [N][K]
  __bf16* wt_ci  = (__bf16*)alloc((size_t)640 * 640  * 2);
  __bf16* wt_qkv = (__bf16*)alloc((size_t)1920 * 640 * 2);
  __bf16* wt_saw = (__bf16*)alloc((size_t)640 * 640  * 2);
  __bf16* wt_caq = (__bf16*)alloc((size_t)640 * 640  * 2);
  __bf16* wt_cak = (__bf16*)alloc((size_t)640 * 512  * 2);
  __bf16* wt_cav = (__bf16*)alloc((size_t)640 * 512  * 2);
  __bf16* wt_caw = (__bf16*)alloc((size_t)640 * 640  * 2);
  __bf16* wt_g1  = (__bf16*)alloc((size_t)5120 * 640 * 2);
  __bf16* wt_g2  = (__bf16*)alloc((size_t)640 * 2560 * 2);
  __bf16* wt_co  = (__bf16*)alloc((size_t)640 * 640  * 2);
  float* bufA = (float*)alloc((size_t)STOK * 640 * 4);   // norm outputs
  float* bufT1= (float*)alloc((size_t)STOK * 640 * 4);   // residual streams
  float* bufT2= (float*)alloc((size_t)STOK * 640 * 4);
  float* bufQK= (float*)alloc((size_t)STOK * 1920 * 4);  // qkv / q2
  float* bufAT= (float*)alloc((size_t)STOK * 640 * 4);   // attention out
  float* bufGG= (float*)alloc((size_t)STOK * 2560 * 4);  // geglu
  float* bufK2= (float*)alloc((size_t)640 * 640 * 4);    // 616 rows padded
  float* bufV2= (float*)alloc((size_t)640 * 640 * 4);

  auto cvtT = [&](const float* s, __bf16* d, int K, int N) {
    dim3 g(K / 32, N / 32);
    k_cvt_t<<<g, dim3(32, 8), 0, stream>>>(s, d, K, N);
  };
  auto gemm = [&](int mode, const float* A, const __bf16* Wt, const float* bias,
                  const float* aux, float* o, int M, int N, int K, int lda,
                  int ldo) {
    dim3 g((M + 63) / 64, N / 128);
    k_gemm<<<g, 256, 0, stream>>>(mode, A, Wt, bias, aux, o, M, N, K, lda, ldo);
  };

  // weights -> bf16, transposed to [N][K]
  cvtT(ci_w,  wt_ci,  640, 640);
  cvtT(qkv_w, wt_qkv, 640, 1920);
  cvtT(saw_w, wt_saw, 640, 640);
  cvtT(caq_w, wt_caq, 640, 640);
  cvtT(cak_w, wt_cak, 512, 640);
  cvtT(cav_w, wt_cav, 512, 640);
  cvtT(caw_w, wt_caw, 640, 640);
  cvtT(g1_w,  wt_g1,  640, 5120);
  cvtT(g2_w,  wt_g2,  2560, 640);
  cvtT(co_w,  wt_co,  640, 640);

  // GroupNorm -> token-major A
  k_groupnorm<<<BDIM * GDIM, 256, 0, stream>>>(x, gn_g, gn_b, bufA);
  // conv_input (1x1): T1 = A @ ci + b
  gemm(0, bufA, wt_ci, ci_b, nullptr, bufT1, STOK, 640, 640, 640, 640);

  // --- self attention ---
  k_layernorm<<<STOK, 256, 0, stream>>>(bufT1, ln1_g, ln1_b, bufA);
  gemm(0, bufA, wt_qkv, qkv_b, nullptr, bufQK, STOK, 1920, 640, 640, 1920);
  {
    dim3 g(HWPIX / FA_BM, BDIM * NHEAD);
    k_flash<<<g, 128, 0, stream>>>(bufQK, bufQK + 640, bufQK + 1280, bufAT,
                                   HWPIX, 1920, 1920, 640, HWPIX, HWPIX,
                                   0.1118033988749895f /* 1/sqrt(80) */);
  }
  gemm(1, bufAT, wt_saw, saw_b, bufT1, bufT2, STOK, 640, 640, 640, 640);

  // --- cross attention ---
  k_layernorm<<<STOK, 256, 0, stream>>>(bufT2, ln2_g, ln2_b, bufA);
  gemm(0, bufA, wt_caq, caq_b, nullptr, bufQK, STOK, 640, 640, 640, 640);
  gemm(0, p, wt_cak, cak_b, nullptr, bufK2, BDIM * PTOK, 640, 512, 512, 640);
  gemm(0, p, wt_cav, cav_b, nullptr, bufV2, BDIM * PTOK, 640, 512, 512, 640);
  {
    dim3 g(HWPIX / FA_BM, BDIM * NHEAD);
    k_flash<<<g, 128, 0, stream>>>(bufQK, bufK2, bufV2, bufAT,
                                   PTOK, 640, 640, 640, HWPIX, PTOK,
                                   0.1118033988749895f);
  }
  gemm(1, bufAT, wt_caw, caw_b, bufT2, bufT1, STOK, 640, 640, 640, 640);

  // --- GEGLU MLP ---
  k_layernorm<<<STOK, 256, 0, stream>>>(bufT1, ln3_g, ln3_b, bufA);
  gemm(0, bufA, wt_g1, g1_b, nullptr, bufGG, STOK, 2560, 640, 640, 2560);
  gemm(2, bufA, wt_g1 + (size_t)2560 * 640, g1_b + 2560, bufGG, bufGG,
       STOK, 2560, 640, 640, 2560);
  gemm(1, bufGG, wt_g2, g2_b, bufT1, bufT2, STOK, 640, 2560, 2560, 640);

  // conv_output (1x1) + long residual, transposed to NCHW
  gemm(3, bufT2, wt_co, co_b, x, out, STOK, 640, 640, 640, 640);
}
